// MoELayer_26182120637040
// MI455X (gfx1250) — matrix operand order repaired and checked by reading, not compiled, not run on hardware
//
#include <hip/hip_runtime.h>
#include <hip/hip_bf16.h>

// ---------------- constants (match reference) ----------------
#define E_    8
#define TOPK_ 2
#define D_    1024
#define DF_   4096
#define HC_   256      // hidden chunk per outer iteration
#define KT_   32       // WMMA K for bf16
#define TILE_M 32      // token rows per block in FFN kernel (2 M-tiles)

typedef __bf16  bf16t;
typedef bf16t          v16bf __attribute__((ext_vector_type(16)));
typedef float          v8f   __attribute__((ext_vector_type(8)));
typedef unsigned short v8u   __attribute__((ext_vector_type(8)));
typedef unsigned short v16u  __attribute__((ext_vector_type(16)));

// float -> bf16 bits, round-to-nearest-even
__device__ __forceinline__ unsigned short f2bf(float f) {
    unsigned int u = __float_as_uint(f);
    u += 0x7FFFu + ((u >> 16) & 1u);
    return (unsigned short)(u >> 16);
}

// 16x32 bf16 A-fragment from an LDS row (row-major).
// lanes 0-15: K = k0+0..7 (elems 0..7), k0+16..23 (elems 8..15)
// lanes 16-31: K = k0+8..15, k0+24..31
__device__ __forceinline__ v16bf load_a_frag(const unsigned short* row, int k0, int khalf) {
    const v8u* p0 = (const v8u*)(row + k0 + khalf * 8);
    const v8u* p1 = (const v8u*)(row + k0 + khalf * 8 + 16);
    v8u a0 = *p0, a1 = *p1;
    v16u au = __builtin_shufflevector(a0, a1, 0,1,2,3,4,5,6,7,8,9,10,11,12,13,14,15);
    return __builtin_bit_cast(v16bf, au);
}

// 32x16 bf16 B-fragment pre-swizzled in global: 32 contiguous bytes per lane.
__device__ __forceinline__ v16bf load_b_global(const unsigned short* p) {
    const v8u* q = (const v8u*)p;
    v8u b0 = q[0], b1 = q[1];
    v16u bu = __builtin_shufflevector(b0, b1, 0,1,2,3,4,5,6,7,8,9,10,11,12,13,14,15);
    return __builtin_bit_cast(v16bf, bu);
}

__device__ __forceinline__ v8f wmma_bf16(v16bf a, v16bf b, v8f c) {
    return __builtin_amdgcn_wmma_f32_16x16x32_bf16(false, a, false, b, (short)0, c, false, false);
}

// ---------------- kernel 0: zero counters ----------------
__global__ void moe_zero_kernel(int* p) {
    if (threadIdx.x < E_) p[threadIdx.x] = 0;
}

// ---------------- kernel 1: routing (one wave per token) ----------------
__global__ __launch_bounds__(256) void moe_route_kernel(
    const float* __restrict__ x, const float* __restrict__ gw,
    int* __restrict__ sel, float* __restrict__ gatew, int* __restrict__ cnt, int T) {
    int wave = threadIdx.x >> 5, lane = threadIdx.x & 31;
    int t = blockIdx.x * 8 + wave;
    if (t >= T) return;
    float lg[E_];
#pragma unroll
    for (int e = 0; e < E_; ++e) lg[e] = 0.f;
    const float* xr = x + (size_t)t * D_;
    for (int d = lane; d < D_; d += 32) {
        float xv = xr[d];
#pragma unroll
        for (int e = 0; e < E_; ++e) lg[e] += xv * gw[e * D_ + d];
    }
#pragma unroll
    for (int m = 16; m > 0; m >>= 1) {
#pragma unroll
        for (int e = 0; e < E_; ++e) lg[e] += __shfl_xor(lg[e], m, 32);
    }
    if (lane == 0) {
        int e0 = 0; float g0 = lg[0];
#pragma unroll
        for (int e = 1; e < E_; ++e) if (lg[e] > g0) { g0 = lg[e]; e0 = e; }
        int e1 = (e0 == 0) ? 1 : 0; float g1 = lg[e1];
#pragma unroll
        for (int e = 0; e < E_; ++e) if (e != e0 && lg[e] > g1) { g1 = lg[e]; e1 = e; }
        float w1 = __expf(g1 - g0);          // softmax over {g0,g1}, g0 >= g1
        float inv = 1.f / (1.f + w1);
        sel[t * 2] = e0;  sel[t * 2 + 1] = e1;
        gatew[t * 2] = inv; gatew[t * 2 + 1] = w1 * inv;
        atomicAdd(&cnt[e0], 1);
        atomicAdd(&cnt[e1], 1);
    }
}

// ---------------- kernel 2: tiny exclusive scan over E ----------------
__global__ void moe_scan_kernel(const int* __restrict__ cnt, int* __restrict__ off,
                                int* __restrict__ cur) {
    if (threadIdx.x == 0) {
        int a = 0;
        for (int e = 0; e < E_; ++e) { off[e] = a; cur[e] = a; a += cnt[e]; }
        off[E_] = a;
    }
}

// ---------------- kernel 3: scatter tokens into expert buckets ----------------
__global__ __launch_bounds__(256) void moe_scatter_kernel(
    const int* __restrict__ sel, const float* __restrict__ gatew,
    int* __restrict__ cur, int* __restrict__ rowtok, float* __restrict__ roww, int T) {
    int t = blockIdx.x * 256 + threadIdx.x;
    if (t >= T) return;
#pragma unroll
    for (int k = 0; k < TOPK_; ++k) {
        int e = sel[t * 2 + k];
        int r = atomicAdd(&cur[e], 1);
        rowtok[r] = t * 2 + k;
        roww[r] = gatew[t * 2 + k];
    }
}

// ---------------- kernel C: convert weights f32 -> bf16 B-fragment layout ----
// src: [E][K][N] row-major f32. dst: per (e, ntile, ktile) a 1KB fragment tile,
// lane l holds 16 contiguous bf16 = B[kt*32 + (l>>4)*8 + j + (j>=8 ? 8 : 0)][nt*16 + (l&15)].
__global__ __launch_bounds__(256) void moe_convw_kernel(
    const float* __restrict__ src, unsigned short* __restrict__ dst, int K, int N) {
    int gid  = blockIdx.x * 256 + threadIdx.x;
    int lane = gid & 31;
    int tile = gid >> 5;
    int KTn = K >> 5, NTn = N >> 4;
    int kt = tile % KTn;
    int rest = tile / KTn;
    int nt = rest % NTn;
    int e  = rest / NTn;
    int col  = nt * 16 + (lane & 15);
    int krow = kt * 32 + (lane >> 4) * 8;
    const float* s = src + ((size_t)e * K + krow) * N + col;
    alignas(16) unsigned short o[16];
#pragma unroll
    for (int j = 0; j < 16; ++j) o[j] = f2bf(s[(size_t)(j + (j >> 3) * 8) * N]);
    v8u* d = (v8u*)(dst + (size_t)gid * 16);
    d[0] = *(const v8u*)&o[0];
    d[1] = *(const v8u*)&o[8];
}

// ---------------- kernel 4: fused expert FFN (WMMA bf16) ----------------
// Block: 256 threads = 8 waves. Tile: 32 token-rows of one expert, full D output.
__global__ __launch_bounds__(256) void moe_ffn_kernel(
    const float* __restrict__ x,
    const unsigned short* __restrict__ W1b, const float* __restrict__ b1,
    const unsigned short* __restrict__ W2b, const float* __restrict__ b2,
    const int* __restrict__ off, const int* __restrict__ rowtok,
    const float* __restrict__ roww, float* __restrict__ out_part,
    int T, int tilesPerE) {

    __shared__ alignas(16) unsigned short xs[TILE_M][D_];     // 64 KB  x tile (bf16)
    __shared__ alignas(16) unsigned short hs[TILE_M][HC_];    // 16 KB  h chunk (bf16)
    __shared__ int   s_tok[TILE_M];
    __shared__ int   s_slot[TILE_M];
    __shared__ float s_w[TILE_M];
    __shared__ int   s_valid[TILE_M];

    int e     = blockIdx.x / tilesPerE;
    int tilei = blockIdx.x % tilesPerE;
    int rbase = off[e] + tilei * TILE_M;
    int rend  = off[e + 1];
    if (rbase >= rend) return;

    int tid = threadIdx.x;
    int wave = tid >> 5, lane = tid & 31;

    if (tid < TILE_M) {
        int r = rbase + tid;
        bool v = r < rend;
        int rt = v ? rowtok[r] : 0;
        s_tok[tid]   = rt >> 1;
        s_slot[tid]  = rt & 1;
        s_w[tid]     = v ? roww[r] : 0.f;
        s_valid[tid] = v ? 1 : 0;
    }
    __syncthreads();

    // stage x rows into LDS as bf16 (invalid rows -> 0)
    for (int i = tid; i < TILE_M * D_; i += 256) {
        int m = i >> 10;            // / D_
        int dc = i & (D_ - 1);
        float v = s_valid[m] ? x[(size_t)s_tok[m] * D_ + dc] : 0.f;
        xs[m][dc] = f2bf(v);
    }
    __syncthreads();

    const int KT1 = D_ >> 5;   // 32  k-tiles in W1 (K = D)
    const int KT2 = DF_ >> 5;  // 128 k-tiles in W2 (K = DF)
    const unsigned short* W1be = W1b + (size_t)e * (DF_ >> 4) * KT1 * 512;
    const unsigned short* W2be = W2b + (size_t)e * (D_  >> 4) * KT2 * 512;
    const float* b1e = b1 + (size_t)e * DF_;
    const float* b2e = b2 + (size_t)e * D_;

    int mrow  = lane & 15;
    int khalf = lane >> 4;

    v8f acc2[8][2];             // [ntile][mtile]: wave owns 128 cols x 32 rows
#pragma unroll
    for (int i = 0; i < 8; ++i) { acc2[i][0] = (v8f)0.0f; acc2[i][1] = (v8f)0.0f; }

    for (int hc0 = 0; hc0 < DF_; hc0 += HC_) {
        // ---- GEMM1: h[32][HC_] = silu(x @ W1[:, hc0:hc0+HC_] + b1) ----
        v8f acc1[2][2];
        acc1[0][0] = (v8f)0.0f; acc1[0][1] = (v8f)0.0f;
        acc1[1][0] = (v8f)0.0f; acc1[1][1] = (v8f)0.0f;
        for (int k0 = 0; k0 < D_; k0 += KT_) {
            v16bf a0 = load_a_frag(&xs[mrow][0],      k0, khalf);
            v16bf a1 = load_a_frag(&xs[16 + mrow][0], k0, khalf);
#pragma unroll
            for (int nt = 0; nt < 2; ++nt) {
                int n0 = hc0 + wave * 32 + nt * 16;
                const unsigned short* fp =
                    W1be + ((size_t)(n0 >> 4) * KT1 + (k0 >> 5)) * 512 + lane * 16;
                v16bf b = load_b_global(fp);
                acc1[nt][0] = wmma_bf16(a0, b, acc1[nt][0]);
                acc1[nt][1] = wmma_bf16(a1, b, acc1[nt][1]);
            }
        }
        // bias + SiLU, write h chunk to LDS as bf16
#pragma unroll
        for (int nt = 0; nt < 2; ++nt) {
            int c = wave * 32 + nt * 16 + mrow;
            float b1v = b1e[hc0 + c];
#pragma unroll
            for (int mt = 0; mt < 2; ++mt) {
                int mb = mt * 16 + 8 * khalf;
#pragma unroll
                for (int r = 0; r < 8; ++r) {
                    float v = acc1[nt][mt][r] + b1v;
                    v = v / (1.f + __expf(-v));
                    hs[mb + r][c] = f2bf(v);
                }
            }
        }
        __syncthreads();

        // ---- GEMM2: out[32][1024] += h @ W2[hc0:hc0+HC_, :] ----
        for (int k0 = 0; k0 < HC_; k0 += KT_) {
            v16bf a0 = load_a_frag(&hs[mrow][0],      k0, khalf);
            v16bf a1 = load_a_frag(&hs[16 + mrow][0], k0, khalf);
            int kk = hc0 + k0;
#pragma unroll
            for (int nt = 0; nt < 8; ++nt) {
                int n0 = wave * 128 + nt * 16;
                const unsigned short* fp =
                    W2be + ((size_t)(n0 >> 4) * KT2 + (kk >> 5)) * 512 + lane * 16;
                v16bf b = load_b_global(fp);
                acc2[nt][0] = wmma_bf16(a0, b, acc2[nt][0]);
                acc2[nt][1] = wmma_bf16(a1, b, acc2[nt][1]);
            }
        }
        __syncthreads();            // hs reused next chunk
    }

    // ---- epilogue: out_part[slot][token][n] = w * (acc + b2[n]) ----
#pragma unroll
    for (int nt = 0; nt < 8; ++nt) {
        int n = wave * 128 + nt * 16 + mrow;
        float b2v = b2e[n];
#pragma unroll
        for (int mt = 0; mt < 2; ++mt) {
            int mb = mt * 16 + 8 * khalf;
#pragma unroll
            for (int r = 0; r < 8; ++r) {
                int m = mb + r;
                if (s_valid[m]) {
                    float v = s_w[m] * (acc2[nt][mt][r] + b2v);
                    out_part[((size_t)s_slot[m] * T + s_tok[m]) * D_ + n] = v;
                }
            }
        }
    }
}

// ---------------- kernel 5: combine the two slot contributions ----------------
__global__ __launch_bounds__(256) void moe_combine_kernel(
    const float* __restrict__ part, float* __restrict__ out, int n) {
    int i = blockIdx.x * 256 + threadIdx.x;
    if (i < n) out[i] = part[i] + part[(size_t)n + i];
}

// ---------------- launcher ----------------
extern "C" void kernel_launch(void* const* d_in, const int* in_sizes, int n_in,
                              void* d_out, int out_size, void* d_ws, size_t ws_size,
                              hipStream_t stream) {
    const float* x  = (const float*)d_in[0];
    const float* gw = (const float*)d_in[1];
    const float* W1 = (const float*)d_in[2];
    const float* b1 = (const float*)d_in[3];
    const float* W2 = (const float*)d_in[4];
    const float* b2 = (const float*)d_in[5];
    float* out = (float*)d_out;

    const int T = in_sizes[0] / D_;          // B*S tokens

    // workspace layout
    int*   iws      = (int*)d_ws;
    int*   p_cnt    = iws;                   // 8
    int*   p_off    = iws + 8;               // 9
    int*   p_cur    = iws + 20;              // 8
    int*   p_sel    = iws + 32;              // 2T
    int*   p_rowtok = p_sel + 2 * T;         // 2T
    float* p_gatew  = (float*)(p_rowtok + 2 * T);  // 2T
    float* p_roww   = p_gatew + 2 * T;             // 2T
    float* p_part   = p_roww + 2 * T;              // 2*T*D floats
    unsigned short* p_W1b = (unsigned short*)(p_part + (size_t)2 * T * D_); // E*D*DF
    unsigned short* p_W2b = p_W1b + (size_t)E_ * D_ * DF_;                  // E*DF*D

    moe_zero_kernel<<<1, 32, 0, stream>>>(p_cnt);
    moe_route_kernel<<<(T + 7) / 8, 256, 0, stream>>>(x, gw, p_sel, p_gatew, p_cnt, T);
    moe_scan_kernel<<<1, 32, 0, stream>>>(p_cnt, p_off, p_cur);
    moe_scatter_kernel<<<(T + 255) / 256, 256, 0, stream>>>(p_sel, p_gatew, p_cur,
                                                            p_rowtok, p_roww, T);
    // weight conversion to WMMA B-fragment layout (bf16)
    {
        int thr1 = E_ * (DF_ >> 4) * (D_ >> 5) * 32;   // W1: K=D, N=DF
        moe_convw_kernel<<<thr1 / 256, 256, 0, stream>>>(W1, p_W1b, D_, DF_);
        int thr2 = E_ * (D_ >> 4) * (DF_ >> 5) * 32;   // W2: K=DF, N=D
        moe_convw_kernel<<<thr2 / 256, 256, 0, stream>>>(W2, p_W2b, DF_, D_);
    }
    int tilesPerE = T / TILE_M;              // worst case: one expert gets every token
    moe_ffn_kernel<<<E_ * tilesPerE, 256, 0, stream>>>(x, p_W1b, b1, p_W2b, b2,
                                                       p_off, p_rowtok, p_roww,
                                                       p_part, T, tilesPerE);
    int nTD = T * D_;
    moe_combine_kernel<<<(nTD + 255) / 256, 256, 0, stream>>>(p_part, out, nTD);
}
